// NeuralLaplaceModel_59691455480128
// MI455X (gfx1250) — matrix operand order repaired and checked by reading, not compiled, not run on hardware
//
#include <hip/hip_runtime.h>
#include <hip/hip_bf16.h>
#include <math.h>

typedef __attribute__((ext_vector_type(16))) _Float16 v16h;
typedef __attribute__((ext_vector_type(8)))  float    v8f;

#define B_SZ       2048
#define ALEN_C     32
#define TP_C       16
#define STATE_C    17
#define ACT_C      6
#define TERMS_C    33
#define HID_C      64
#define GRUH_C     32
#define OUTD_C     17
#define LAT_C      19
#define ALPHA_C    0.001f
#define NLOGTOL_C  4.6051701859880914f   /* -log(0.01) */
#define PI_C       3.14159265358979323846f

// ---------------------------------------------------------------------------
// Fast math helpers: v_rcp_f32-based, no IEEE div chains.
// ---------------------------------------------------------------------------
__device__ __forceinline__ float frcp(float x) { return __builtin_amdgcn_rcpf(x); }

// tanh(x) = 1 - 2/(e^{2x}+1); saturates correctly for |x| large.
__device__ __forceinline__ float fast_tanh(float x) {
  const float e = __expf(2.0f * x);
  return 1.0f - 2.0f * frcp(e + 1.0f);
}

__device__ __forceinline__ float sigm(float x) {
  return frcp(1.0f + __expf(-x));
}

// ---------------------------------------------------------------------------
// WMMA helpers (CDNA5 wave32, v_wmma_f32_16x16x32_f16)
// ---------------------------------------------------------------------------
__device__ __forceinline__ v8f wmma_f16(v16h a, v16h b, v8f c) {
  return __builtin_amdgcn_wmma_f32_16x16x32_f16(
      /*neg_a=*/false, a, /*neg_b=*/false, b,
      /*c_mod=*/(short)0, c, /*reuse_a=*/false, /*reuse_b=*/false);
}

// A-fragment (16x32 f16) from row-major f16 matrix in LDS.
// Lanes 0-15 hold row M=lane&15, K elems kb..kb+15 in elements 0..15,
// kb = 0 (lanes<16) or 16 (lanes>=16).
__device__ __forceinline__ v16h load_a(const _Float16* base, int ldm, int lane) {
  const int row = lane & 15;
  const int kb  = (lane & 16) ? 16 : 0;
  v16h a;
#pragma unroll
  for (int i = 0; i < 16; ++i) {
    a[i] = base[row * ldm + kb + i];
  }
  return a;
}

// B-fragment (32x16 f16) built from a row-major (N,K) f32 weight matrix in
// global memory: B[k][n] = W[n][k]. Lanes 0-15 hold col n = lane&15 with
// K = 0..15 in elements 0..15; lanes 16-31 hold K = 16..31.
// Kmax clamps (zero pad); Kmax==0 -> all-zero fragment.
__device__ __forceinline__ v16h load_b_global(const float* __restrict__ W,
                                              int ldk, int n, int kc, int Kmax,
                                              int lane) {
  const int klo = (lane & 16) ? 16 : 0;
  v16h b;
#pragma unroll
  for (int i = 0; i < 16; ++i) {
    const int k = kc * 32 + klo + i;
    b[i] = (k < Kmax) ? (_Float16)W[n * ldk + k] : (_Float16)0.0f;
  }
  return b;
}

// ---------------------------------------------------------------------------
// Kernel 1: 2-layer GRU (seq 32, H=32) + encoder head -> p = [obs, p_action]
// One wave handles 16 batch rows; all matmuls via WMMA (K padded to 32).
// ---------------------------------------------------------------------------
#define GRU_WAVES 2

__global__ __launch_bounds__(GRU_WAVES * 32)
void gru_kernel(const float* __restrict__ obs, const float* __restrict__ action,
                const float* __restrict__ Wih0, const float* __restrict__ Whh0,
                const float* __restrict__ bih0, const float* __restrict__ bhh0,
                const float* __restrict__ Wih1, const float* __restrict__ Whh1,
                const float* __restrict__ bih1, const float* __restrict__ bhh1,
                const float* __restrict__ Wout, const float* __restrict__ bout,
                float* __restrict__ p_out)
{
  // Weight B-fragments staged in fragment order: mat(4) x ntile(6) x lane(32) x 16 halves
  __shared__ __align__(32) _Float16 wfrag[4 * 6 * 32 * 16];
  __shared__ __align__(32) _Float16 hbuf[GRU_WAVES][2][16 * GRUH_C];

  const int tid = threadIdx.x;
  {
    const float* mats[4]   = {Wih0, Whh0, Wih1, Whh1};
    const int    rowlen[4] = {ACT_C, GRUH_C, GRUH_C, GRUH_C};
#pragma unroll 1
    for (int c = tid; c < 4 * 6 * 32; c += blockDim.x) {
      const int m = c / 192, rem = c % 192, nt = rem / 32, l = rem % 32;
      const int n   = nt * 16 + (l & 15);
      const int klo = (l & 16) ? 16 : 0;
      const int rl  = rowlen[m];
      const float* W = mats[m];
      _Float16* dst = &wfrag[c * 16];
#pragma unroll
      for (int i = 0; i < 16; ++i) {
        const int k = klo + i;
        dst[i] = (k < rl) ? (_Float16)W[n * rl + k] : (_Float16)0.0f;
      }
    }
  }
  __syncthreads();

  const int lane = tid & 31;
  const int wv   = tid >> 5;
  const int gb0  = blockIdx.x * (GRU_WAVES * 16) + wv * 16;
  const int col  = lane & 15;
  const int rofs = (lane & 16) ? 8 : 0;

  _Float16* h0b = hbuf[wv][0];
  _Float16* h1b = hbuf[wv][1];
  for (int i = lane; i < 16 * GRUH_C; i += 32) { h0b[i] = (_Float16)0.0f; h1b[i] = (_Float16)0.0f; }

  // Per-lane gate biases (column = ht*16+col of each 32-wide gate block)
  float bI0[2][3], bH0[2][3], bI1[2][3], bH1[2][3];
#pragma unroll
  for (int ht = 0; ht < 2; ++ht)
#pragma unroll
    for (int g = 0; g < 3; ++g) {
      const int j = g * GRUH_C + ht * 16 + col;
      bI0[ht][g] = bih0[j]; bH0[ht][g] = bhh0[j];
      bI1[ht][g] = bih1[j]; bH1[ht][g] = bhh1[j];
    }

  float h0c[2][8], h1c[2][8];
#pragma unroll
  for (int ht = 0; ht < 2; ++ht)
#pragma unroll
    for (int r = 0; r < 8; ++r) { h0c[ht][r] = 0.0f; h1c[ht][r] = 0.0f; }

  const _Float16* wbase = wfrag;
#define WFRAG(m, nt) (*(const v16h*)&wbase[(((m) * 6 + (nt)) * 32 + lane) * 16])

#pragma unroll 1
  for (int t = 0; t < ALEN_C; ++t) {
    const int at = ALEN_C - 1 - t;  // time reversal

    // x_t A-fragment (K padded 6 -> 32)
    v16h xa;
#pragma unroll
    for (int i = 0; i < 16; ++i) xa[i] = (_Float16)0.0f;
    if (lane < 16) {
      const float* ap = &action[((size_t)(gb0 + col) * ALEN_C + at) * ACT_C];
#pragma unroll
      for (int i = 0; i < ACT_C; ++i) xa[i] = (_Float16)(ap[i] * (1.0f / 3.0f));
    }

    v8f xw[6], gh[6];
    const v8f zacc = {0.f, 0.f, 0.f, 0.f, 0.f, 0.f, 0.f, 0.f};

    // ---- layer 0 ----
    v16h h0A = load_a(h0b, GRUH_C, lane);
#pragma unroll
    for (int g = 0; g < 6; ++g) {
      xw[g] = wmma_f16(xa,  WFRAG(0, g), zacc);
      gh[g] = wmma_f16(h0A, WFRAG(1, g), zacc);
    }
#pragma unroll
    for (int ht = 0; ht < 2; ++ht)
#pragma unroll
      for (int r = 0; r < 8; ++r) {
        const float xr = xw[ht][r]     + bI0[ht][0], hr = gh[ht][r]     + bH0[ht][0];
        const float xz = xw[2 + ht][r] + bI0[ht][1], hz = gh[2 + ht][r] + bH0[ht][1];
        const float xn = xw[4 + ht][r] + bI0[ht][2], hn = gh[4 + ht][r] + bH0[ht][2];
        const float rg = sigm(xr + hr);
        const float zg = sigm(xz + hz);
        const float ng = fast_tanh(xn + rg * hn);
        const float h  = (1.0f - zg) * ng + zg * h0c[ht][r];
        h0c[ht][r] = h;
        h0b[(r + rofs) * GRUH_C + ht * 16 + col] = (_Float16)h;
      }

    // ---- layer 1 (input = fresh h0) ----
    v16h h0n = load_a(h0b, GRUH_C, lane);
    v16h h1A = load_a(h1b, GRUH_C, lane);
#pragma unroll
    for (int g = 0; g < 6; ++g) {
      xw[g] = wmma_f16(h0n, WFRAG(2, g), zacc);
      gh[g] = wmma_f16(h1A, WFRAG(3, g), zacc);
    }
#pragma unroll
    for (int ht = 0; ht < 2; ++ht)
#pragma unroll
      for (int r = 0; r < 8; ++r) {
        const float xr = xw[ht][r]     + bI1[ht][0], hr = gh[ht][r]     + bH1[ht][0];
        const float xz = xw[2 + ht][r] + bI1[ht][1], hz = gh[2 + ht][r] + bH1[ht][1];
        const float xn = xw[4 + ht][r] + bI1[ht][2], hn = gh[4 + ht][r] + bH1[ht][2];
        const float rg = sigm(xr + hr);
        const float zg = sigm(xz + hz);
        const float ng = fast_tanh(xn + rg * hn);
        const float h  = (1.0f - zg) * ng + zg * h1c[ht][r];
        h1c[ht][r] = h;
        h1b[(r + rofs) * GRUH_C + ht * 16 + col] = (_Float16)h;
      }
  }
#undef WFRAG

  // encoder head: p_action = h2_last @ Wout.T + bout  (2 outputs)
  if (lane < 16) {
    const int gb = gb0 + lane;
    float a0 = bout[0], a1 = bout[1];
#pragma unroll
    for (int j = 0; j < GRUH_C; ++j) {
      const float hv = (float)h1b[lane * GRUH_C + j];
      a0 += hv * Wout[j];
      a1 += hv * Wout[GRUH_C + j];
    }
    p_out[(size_t)gb * LAT_C + STATE_C]     = a0;
    p_out[(size_t)gb * LAT_C + STATE_C + 1] = a1;
  }
#pragma unroll 1
  for (int idx = lane; idx < 16 * STATE_C; idx += 32) {
    const int r = idx / STATE_C, c = idx % STATE_C;
    p_out[(size_t)(gb0 + r) * LAT_C + c] = obs[(size_t)(gb0 + r) * STATE_C + c];
  }
}

// ---------------------------------------------------------------------------
// Kernel 2: Laplace features -> MLP (85->64->64->1122 via WMMA) -> series.
// One wave per 16 rows (row = b*TP + tp). W3 columns permuted so (theta,phi)
// partners are adjacent; ang = pi*k/2 collapses cos/sin to {0,+-1}.
// ---------------------------------------------------------------------------
__global__ __launch_bounds__(32)
void laplace_mlp_kernel(const float* __restrict__ ts, const float* __restrict__ p,
                        const float* __restrict__ W1, const float* __restrict__ b1,
                        const float* __restrict__ W2, const float* __restrict__ b2,
                        const float* __restrict__ W3, const float* __restrict__ b3,
                        float* __restrict__ out)
{
  extern __shared__ char smem[];
  float*    tval = (float*)smem;                       // 16
  float*    outb = tval + 16;                          // 16*1136
  _Float16* feat = (_Float16*)(outb + 16 * 1136);      // 16*96
  _Float16* h1b  = feat + 16 * 96;                     // 16*64
  _Float16* h2b  = h1b + 16 * 64;                      // 16*64

  const int lane = threadIdx.x;
  const int R0   = blockIdx.x * 16;
  const int col  = lane & 15;
  const int rofs = (lane & 16) ? 8 : 0;
  const v8f zacc = {0.f, 0.f, 0.f, 0.f, 0.f, 0.f, 0.f, 0.f};

  if (lane < 16) tval[lane] = ts[R0 + lane];   // ts[b][tp] flat == ts[R]

  // Features: [theta_s(33) | phi_s(33) | p(19) | pad(11)]
#pragma unroll 1
  for (int idx = lane; idx < 16 * 96; idx += 32) {
    const int r = idx / 96, c = idx % 96;
    const float t     = tval[r];
    const float rTc   = frcp(2.0f * t);
    const float gamma = ALPHA_C + NLOGTOL_C * rTc;
    float v = 0.0f;
    if (c < TERMS_C) {
      const float sim = PI_C * (float)c * rTc;
      v = atan2f(sim, gamma);
    } else if (c < 2 * TERMS_C) {
      const int   k   = c - TERMS_C;
      const float sim = PI_C * (float)k * rTc;
      const float sq  = gamma * gamma + sim * sim;
      v = asinf((sq - 1.0f) * frcp(sq + 1.0f));
    } else if (c < 2 * TERMS_C + LAT_C) {
      const int b = (R0 + r) / TP_C;
      v = p[(size_t)b * LAT_C + (c - 2 * TERMS_C)];
    }
    feat[r * 96 + c] = (_Float16)v;
  }

  // ---- MLP layer 1: (16x96) @ (96x64), K actual 85 ----
  {
    const v16h a0 = load_a(feat,      96, lane);
    const v16h a1 = load_a(feat + 32, 96, lane);
    const v16h a2 = load_a(feat + 64, 96, lane);
#pragma unroll
    for (int nt = 0; nt < 4; ++nt) {
      const int n = nt * 16 + col;
      v8f acc = zacc;
      acc = wmma_f16(a0, load_b_global(W1, 85, n, 0, 85, lane), acc);
      acc = wmma_f16(a1, load_b_global(W1, 85, n, 1, 85, lane), acc);
      acc = wmma_f16(a2, load_b_global(W1, 85, n, 2, 85, lane), acc);
      const float bias = b1[n];
#pragma unroll
      for (int r = 0; r < 8; ++r)
        h1b[(r + rofs) * HID_C + n] = (_Float16)fast_tanh(acc[r] + bias);
    }
  }

  // ---- MLP layer 2: (16x64) @ (64x64) ----
  {
    const v16h a0 = load_a(h1b,      64, lane);
    const v16h a1 = load_a(h1b + 32, 64, lane);
#pragma unroll
    for (int nt = 0; nt < 4; ++nt) {
      const int n = nt * 16 + col;
      v8f acc = zacc;
      acc = wmma_f16(a0, load_b_global(W2, 64, n, 0, 64, lane), acc);
      acc = wmma_f16(a1, load_b_global(W2, 64, n, 1, 64, lane), acc);
      const float bias = b2[n];
#pragma unroll
      for (int r = 0; r < 8; ++r)
        h2b[(r + rofs) * HID_C + n] = (_Float16)fast_tanh(acc[r] + bias);
    }
  }

  // ---- MLP layer 3: (16x64) @ (64x1122), columns permuted pair-wise.
  // permuted col 2p   -> orig col p        (theta raw, p = m*33+k, m<17)
  // permuted col 2p+1 -> orig col p + 561  (phi raw)
  {
    const v16h a0 = load_a(h2b,      64, lane);
    const v16h a1 = load_a(h2b + 32, 64, lane);
#pragma unroll 1
    for (int nt = 0; nt < 71; ++nt) {
      const int  colp  = nt * 16 + col;
      const int  pidx  = colp >> 1;
      const int  orig  = (colp & 1) ? (pidx + 561) : pidx;
      const bool valid = colp < 1122;
      v8f acc = zacc;
      acc = wmma_f16(a0, load_b_global(W3, 64, valid ? orig : 0, 0, valid ? 64 : 0, lane), acc);
      acc = wmma_f16(a1, load_b_global(W3, 64, valid ? orig : 0, 1, valid ? 64 : 0, lane), acc);
      if (valid) {
        const float bias = b3[orig];
#pragma unroll
        for (int r = 0; r < 8; ++r)
          outb[(r + rofs) * 1136 + colp] = acc[r] + bias;
      }
    }
  }

  // ---- series reduction: ang = pi*k/2 -> cos/sin in {0,+-1} ----
#pragma unroll 1
  for (int w = lane; w < 16 * OUTD_C; w += 32) {
    const int   r = w / OUTD_C, m = w % OUTD_C;
    const float t     = tval[r];
    const float rTc   = frcp(2.0f * t);
    const float gamma = ALPHA_C + NLOGTOL_C * rTc;
    const float coef  = __expf(gamma * t) * rTc;
    float sum = 0.0f;
#pragma unroll 1
    for (int k = 0; k < TERMS_C; ++k) {
      const int   pp    = m * TERMS_C + k;
      const float thr   = outb[r * 1136 + 2 * pp];
      const float phr   = outb[r * 1136 + 2 * pp + 1];
      const float theta = fast_tanh(thr) * PI_C;
      const float phi   = fast_tanh(phr) * (PI_C * 0.5f);
      const float base  = __cosf(phi) * frcp(1.0f - __sinf(phi));
      float c;
      switch (k & 3) {
        case 0:  c =  __cosf(theta) * base; break;
        case 1:  c = -__sinf(theta) * base; break;
        case 2:  c = -__cosf(theta) * base; break;
        default: c =  __sinf(theta) * base; break;
      }
      if (k == 0) c *= 0.5f;
      sum += c;
    }
    out[(size_t)(R0 + r) * OUTD_C + m] = coef * sum;
  }
}

// ---------------------------------------------------------------------------
extern "C" void kernel_launch(void* const* d_in, const int* in_sizes, int n_in,
                              void* d_out, int out_size, void* d_ws, size_t ws_size,
                              hipStream_t stream) {
  (void)in_sizes; (void)n_in; (void)out_size; (void)ws_size;
  const float* obs    = (const float*)d_in[0];
  const float* action = (const float*)d_in[1];
  const float* ts     = (const float*)d_in[2];
  const float* Wih0   = (const float*)d_in[3];
  const float* Whh0   = (const float*)d_in[4];
  const float* bih0   = (const float*)d_in[5];
  const float* bhh0   = (const float*)d_in[6];
  const float* Wih1   = (const float*)d_in[7];
  const float* Whh1   = (const float*)d_in[8];
  const float* bih1   = (const float*)d_in[9];
  const float* bhh1   = (const float*)d_in[10];
  const float* Wout   = (const float*)d_in[11];
  const float* bout   = (const float*)d_in[12];
  const float* W1     = (const float*)d_in[13];
  const float* b1     = (const float*)d_in[14];
  const float* W2     = (const float*)d_in[15];
  const float* b2     = (const float*)d_in[16];
  const float* W3     = (const float*)d_in[17];
  const float* b3     = (const float*)d_in[18];
  float*       outp   = (float*)d_out;
  float*       p_ws   = (float*)d_ws;  // B x 19 latent vectors

  const int gru_blocks = B_SZ / (GRU_WAVES * 16);
  gru_kernel<<<gru_blocks, GRU_WAVES * 32, 0, stream>>>(
      obs, action, Wih0, Whh0, bih0, bhh0, Wih1, Whh1, bih1, bhh1,
      Wout, bout, p_ws);

  const size_t smem2 = (16 + 16 * 1136) * sizeof(float) +
                       (16 * 96 + 16 * 64 + 16 * 64) * sizeof(_Float16);
  const int mlp_blocks = (B_SZ * TP_C) / 16;
  laplace_mlp_kernel<<<mlp_blocks, 32, smem2, stream>>>(
      ts, p_ws, W1, b1, W2, b2, W3, b3, outp);
}
